// RNN_2594160247281
// MI455X (gfx1250) — matrix operand order repaired and checked by reading, not compiled, not run on hardware
//
#include <hip/hip_runtime.h>
#include <math.h>

typedef __bf16 bf16;
typedef __attribute__((ext_vector_type(16))) __bf16 v16bf;
typedef __attribute__((ext_vector_type(8)))  __bf16 v8bf;
typedef __attribute__((ext_vector_type(4)))  __bf16 v4bf;
typedef __attribute__((ext_vector_type(8)))  float  v8f;
typedef __attribute__((ext_vector_type(4)))  float  v4f;

#define T_STEPS 512
#define B_SZ    128
#define D_SZ    256
#define H_SZ    512

__device__ __forceinline__ v4bf cvt4(v4f f) {
    v4bf r;
    r[0] = (bf16)f[0]; r[1] = (bf16)f[1]; r[2] = (bf16)f[2]; r[3] = (bf16)f[3];
    return r;
}

// ---------------------------------------------------------------------------
// Kernel 1: xin = x @ W_in^T + b_in     (M=T*B=65536, N=H=512, K=D=256)
// Block computes a 32(M) x 64(N) tile. 256 threads = 8 waves, 1 WMMA tile/wave.
// x is streamed with NT loads (67 MB, read once); xin stores stay RT so the
// 134 MB intermediate remains resident in the 192 MB L2 for the scan kernel.
// ---------------------------------------------------------------------------
__global__ __launch_bounds__(256) void rnn_xin_gemm(
    const float* __restrict__ x, const float* __restrict__ W_in,
    const float* __restrict__ b_in, float* __restrict__ xin_out)
{
    __shared__ __align__(16) bf16 xs[32][264];   // x tile, bf16, 528B stride (16B mult)
    __shared__ __align__(16) bf16 ws[64][264];   // W tile, bf16 (row n, contiguous k)

    const int tid  = threadIdx.x;
    const int lane = tid & 31;
    const int wave = tid >> 5;
    const int m0   = blockIdx.y * 32;   // rows over T*B
    const int n0   = blockIdx.x * 64;   // cols over H

    // stage + convert x tile (32x256) : 16B NT loads -> packed 4xbf16 LDS stores
    for (int i = 0; i < 8; ++i) {
        int idx = (i * 256 + tid) * 4;              // 0..8188
        int r = idx >> 8, d = idx & 255;
        v4f f = __builtin_nontemporal_load(
            (const v4f*)&x[(size_t)(m0 + r) * D_SZ + d]);
        *(v4bf*)&xs[r][d] = cvt4(f);
    }
    // stage + convert W_in tile (64x256) : W_in is tiny & L2-hot, RT loads
    for (int i = 0; i < 16; ++i) {
        int idx = (i * 256 + tid) * 4;              // 0..16380
        int r = idx >> 8, d = idx & 255;
        v4f f = *(const v4f*)&W_in[(size_t)(n0 + r) * D_SZ + d];
        *(v4bf*)&ws[r][d] = cvt4(f);
    }
    __syncthreads();

    const int mt   = wave >> 2;              // 0..1
    const int nt   = wave & 3;               // 0..3
    const int bl   = lane & 15;
    const int arow = mt * 16 + bl;           // A: lane holds row l%16
    const int akb  = (lane >> 4) * 8;        // A: halves [0..7]=K akb.., [8..15]=K akb+16..
    const int bkb  = (lane >> 4) * 8;        // B: halves [0..7]=K bkb.., [8..15]=K bkb+16..
    const int brow = nt * 16 + bl;           // B: lane holds col l%16 (= W row)

    v8f acc;
    for (int v = 0; v < 8; ++v) acc[v] = 0.0f;

    for (int ks = 0; ks < D_SZ / 32; ++ks) {
        int kb = ks * 32;
        union { v16bf v; v8bf h[2]; } a, b;
        a.h[0] = *(const v8bf*)&xs[arow][kb + akb];
        a.h[1] = *(const v8bf*)&xs[arow][kb + akb + 16];
        b.h[0] = *(const v8bf*)&ws[brow][kb + bkb];
        b.h[1] = *(const v8bf*)&ws[brow][kb + bkb + 16];
        acc = __builtin_amdgcn_wmma_f32_16x16x32_bf16(
                  false, a.v, false, b.v, (short)0, acc, false, false);
    }

    // C/D layout: lane l, vgpr v -> (m = v + 8*(l/16), n = l%16)
    const int mrow = m0 + mt * 16 + ((lane >> 4) * 8);
    const int col  = n0 + nt * 16 + bl;
    const float bias = b_in[col];
    for (int v = 0; v < 8; ++v)
        xin_out[(size_t)(mrow + v) * H_SZ + col] = acc[v] + bias;  // RT: keep in L2
}

// ---------------------------------------------------------------------------
// Init: convert provided h0 (fp32) -> bf16 ping buffer, reset sync counter.
// ---------------------------------------------------------------------------
__global__ void rnn_scan_init(const float* __restrict__ h,
                              bf16* __restrict__ h0,
                              unsigned int* __restrict__ sync_cnt)
{
    int i = blockIdx.x * blockDim.x + threadIdx.x;
    if (i < B_SZ * H_SZ) h0[i] = (bf16)h[i];
    if (i == 0) *sync_cnt = 0u;
}

// ---------------------------------------------------------------------------
// Kernel 2: persistent recurrent scan.
// 16 workgroups, each owns a 32-wide slice of H; W_lat slice lives in LDS
// (bf16, 32x512 = 32KB) for all 512 steps. h ping-pongs through a bf16
// global buffer (L2-resident); device-scope atomic barrier per timestep.
// out[] holds xin on entry (L2-hot from kernel 1); final h states are stored
// with NT hints (never re-read) so L2 stays reserved for xin + h buffers.
// ---------------------------------------------------------------------------
__global__ __launch_bounds__(256) void rnn_scan(
    const float* __restrict__ W_lat, const float* __restrict__ b_lat,
    float* __restrict__ out, bf16* __restrict__ h0buf, bf16* __restrict__ h1buf,
    unsigned int* __restrict__ sync_cnt, int n_wg)
{
    __shared__ __align__(16) bf16 wl[32][520];   // W_lat slice, 1040B stride (16B mult)

    const int tid  = threadIdx.x;
    const int lane = tid & 31;
    const int wave = tid >> 5;
    const int n0   = blockIdx.x * 32;

    // stage + convert W_lat slice: rows n0..n0+31, K = 512 each (16384 elems)
    for (int i = 0; i < 16; ++i) {
        int idx = (i * 256 + tid) * 4;              // 0..16380
        int r = idx >> 9, k = idx & 511;
        v4f f = *(const v4f*)&W_lat[(size_t)(n0 + r) * H_SZ + k];
        *(v4bf*)&wl[r][k] = cvt4(f);
    }
    __syncthreads();

    const int bl    = lane & 15;
    const int arow  = wave * 16 + bl;            // batch row (M = B = 128, wave = mtile)
    const int akb   = (lane >> 4) * 8;
    const int bkb   = (lane >> 4) * 8;
    const int mbase = wave * 16 + ((lane >> 4) * 8);

    float biasv[2];
    for (int nt = 0; nt < 2; ++nt) biasv[nt] = b_lat[n0 + nt * 16 + bl];

    bf16* hb[2] = { h0buf, h1buf };

    for (int t = 0; t < T_STEPS; ++t) {
        const bf16* __restrict__ hprev = hb[t & 1];
        bf16* __restrict__ hnext       = hb[(t + 1) & 1];

        v8f acc[2];
        for (int nt = 0; nt < 2; ++nt)
            for (int v = 0; v < 8; ++v) acc[nt][v] = 0.0f;

        const bf16* __restrict__ arowp = hprev + (size_t)arow * H_SZ;
        for (int ks = 0; ks < H_SZ / 32; ++ks) {
            int kb = ks * 32;
            union { v16bf v; v8bf h[2]; } a;
            a.h[0] = *(const v8bf*)(arowp + kb + akb);
            a.h[1] = *(const v8bf*)(arowp + kb + akb + 16);
            for (int nt = 0; nt < 2; ++nt) {
                union { v16bf v; v8bf h[2]; } b;
                int brow = nt * 16 + bl;
                b.h[0] = *(const v8bf*)&wl[brow][kb + bkb];
                b.h[1] = *(const v8bf*)&wl[brow][kb + bkb + 16];
                acc[nt] = __builtin_amdgcn_wmma_f32_16x16x32_bf16(
                              false, a.v, false, b.v, (short)0, acc[nt], false, false);
            }
        }

        // epilogue: h = tanh(acc + xin + b_lat); NT fp32 to out, RT bf16 to hnext
        float* outt = out + (size_t)t * B_SZ * H_SZ;
        for (int nt = 0; nt < 2; ++nt) {
            int col = n0 + nt * 16 + bl;
            for (int v = 0; v < 8; ++v) {
                size_t idx = (size_t)(mbase + v) * H_SZ + col;
                float r = tanhf(acc[nt][v] + outt[idx] + biasv[nt]);
                __builtin_nontemporal_store(r, &outt[idx]);   // final output, not re-read
                hnext[idx] = (bf16)r;                         // RT: re-read next step
            }
        }

        // device-wide barrier: all 16 WGs must finish step t before t+1
        __threadfence();
        __syncthreads();
        if (tid == 0) {
            __hip_atomic_fetch_add(sync_cnt, 1u, __ATOMIC_ACQ_REL,
                                   __HIP_MEMORY_SCOPE_AGENT);
            unsigned target = (unsigned)n_wg * (unsigned)(t + 1);
            while (__hip_atomic_load(sync_cnt, __ATOMIC_ACQUIRE,
                                     __HIP_MEMORY_SCOPE_AGENT) < target)
                __builtin_amdgcn_s_sleep(2);
        }
        __syncthreads();
    }
}

// ---------------------------------------------------------------------------
extern "C" void kernel_launch(void* const* d_in, const int* in_sizes, int n_in,
                              void* d_out, int out_size, void* d_ws, size_t ws_size,
                              hipStream_t stream)
{
    const float* x     = (const float*)d_in[0];
    const float* h0f   = (const float*)d_in[1];
    const float* W_in  = (const float*)d_in[2];
    const float* b_in  = (const float*)d_in[3];
    const float* W_lat = (const float*)d_in[4];
    const float* b_lat = (const float*)d_in[5];
    float* out = (float*)d_out;

    // workspace layout: [0,128KB) h ping (bf16), [128KB,256KB) h pong, then counter
    bf16* hping = (bf16*)d_ws;
    bf16* hpong = hping + (size_t)B_SZ * H_SZ;
    unsigned int* cnt = (unsigned int*)((char*)d_ws + 2u * B_SZ * H_SZ * sizeof(bf16));

    rnn_scan_init<<<(B_SZ * H_SZ + 255) / 256, 256, 0, stream>>>(h0f, hping, cnt);

    // xin -> d_out (in place): grid = (H/64, (T*B)/32)
    rnn_xin_gemm<<<dim3(H_SZ / 64, (T_STEPS * B_SZ) / 32), 256, 0, stream>>>(
        x, W_in, b_in, out);

    // persistent scan: 16 WGs, each owns 32 columns of H
    rnn_scan<<<H_SZ / 32, 256, 0, stream>>>(W_lat, b_lat, out, hping, hpong, cnt, H_SZ / 32);
}
